// PoseSeqModel_19868518711664
// MI455X (gfx1250) — matrix operand order repaired and checked by reading, not compiled, not run on hardware
//
#include <hip/hip_runtime.h>

// PoseSeq GRU decoder for MI455X (gfx1250, wave32, WMMA bf16).
// Strategy:
//  * Convert all weights fp32->bf16 once (they then live in the 192MB L2 for
//    the whole 256-step scan; HBM traffic is negligible after step 0).
//  * Precompute gi_ctx = context @ w_ih0[:,150:].T + b_ih0 (loop-invariant),
//    so the per-step input-gate GEMM has K=160 instead of 918.
//  * Per GRU layer, one fused kernel: 256 threads = 8 wave32s, each wave owns
//    a 16x16 h-tile and accumulates all 6 gate tiles (ir,iz,in,hr,hz,hn) with
//    v_wmma_f32_16x16x32_bf16, then does the sigmoid/tanh gate combine in
//    registers and writes h_new (f32 + bf16). No gate buffers in memory.
//  * Hidden state ping-pongs between two buffers per step (no WAR races).

#define E_DIM 768
#define H_DIM 1024
#define P_DIM 150
#define PPAD  160
#define B_DIM 128
#define LMAX  256
#define G3    3072   // 3*H

typedef __bf16 bf16_t;
typedef __attribute__((ext_vector_type(16))) __bf16 v16bf;
typedef __attribute__((ext_vector_type(8)))  float  v8f;

__device__ __forceinline__ v8f wmma_bf16(v16bf a, v16bf b, v8f c) {
  // D = A(16x32) x B(32x16) + C(16x16 f32)
  return __builtin_amdgcn_wmma_f32_16x16x32_bf16(false, a, false, b, (short)0, c,
                                                 false, false);
}

// A-fragment (16x32 bf16, row-major, leading dim lda):
// lane<16: rows 0-15, K 0..15 ; lane>=16: same rows, K 16..31 (contiguous)
__device__ __forceinline__ v16bf load_frag_a(const bf16_t* __restrict__ A, int lda,
                                             int m0, int k0, int lane) {
  int row = m0 + (lane & 15);
  int kh  = (lane >> 4) << 4;            // 0 or 16
  const bf16_t* p = A + (size_t)row * lda + k0 + kh;
  union { uint4 u[2]; v16bf v; } f;
  f.u[0] = *(const uint4*)(p);           // 8 bf16
  f.u[1] = *(const uint4*)(p + 8);       // 8 bf16, +16B
  return f.v;
}

// B-fragment (32x16 bf16). B[k][n] = W[j0+n][k] with W row-major (N x K).
// lane<16: col n=lane, K 0..15 ; lane>=16: col n=lane-16, K 16..31 (contiguous)
__device__ __forceinline__ v16bf load_frag_b(const bf16_t* __restrict__ W, int ldw,
                                             int j0, int k0, int lane) {
  int row = j0 + (lane & 15);
  int kh  = (lane >> 4) << 4;            // 0 or 16
  const bf16_t* p = W + (size_t)row * ldw + k0 + kh;
  union { uint4 u[2]; v16bf v; } f;
  f.u[0] = *(const uint4*)(p);
  f.u[1] = *(const uint4*)(p + 8);
  return f.v;
}

__device__ __forceinline__ float fsigmoid(float x) {
  return 1.0f / (1.0f + __expf(-x));
}
__device__ __forceinline__ float ftanh(float x) {
  float e = __expf(2.0f * x);
  return 1.0f - 2.0f / (e + 1.0f);       // saturates correctly at +-1
}

// ---------------- fused GRU layer -------------------------------------------
// grid = 64 (column tiles of H), block = 256 (8 waves = 8 row tiles of M=128)
__global__ __launch_bounds__(256) void gru_layer_kernel(
    const bf16_t* __restrict__ x_bf, int Kx,        // (128 x Kx) layer input
    const bf16_t* __restrict__ Wih,                 // (3072 x Kx) bf16
    const float*  __restrict__ gi_init,             // (128x3072) or nullptr
    const float*  __restrict__ b_ih,                // used iff gi_init==nullptr
    const bf16_t* __restrict__ h_bf,                // (128x1024) prev hidden bf16
    const float*  __restrict__ h_f,                 // (128x1024) prev hidden f32
    const bf16_t* __restrict__ Whh,                 // (3072x1024) bf16
    const float*  __restrict__ b_hh,                // (3072)
    float*  __restrict__ hnew_f,                    // (128x1024)
    bf16_t* __restrict__ hnew_bf)                   // (128x1024)
{
  int lane = threadIdx.x & 31;
  int wave = threadIdx.x >> 5;
  int m0   = wave * 16;
  int j0   = blockIdx.x * 16;
  int n    = lane & 15;
  int mrow = m0 + ((lane >> 4) << 3);   // C/D layout: lane>=16 -> M+8

  v8f air, aiz, ain, ahr, ahz, ahn;
  if (gi_init) {
    for (int v = 0; v < 8; ++v) {
      const float* g = gi_init + (size_t)(mrow + v) * G3;
      air[v] = g[j0 + n];
      aiz[v] = g[H_DIM + j0 + n];
      ain[v] = g[2 * H_DIM + j0 + n];
    }
  } else {
    float br = b_ih[j0 + n], bz = b_ih[H_DIM + j0 + n], bn = b_ih[2 * H_DIM + j0 + n];
    for (int v = 0; v < 8; ++v) { air[v] = br; aiz[v] = bz; ain[v] = bn; }
  }
  {
    float br = b_hh[j0 + n], bz = b_hh[H_DIM + j0 + n], bn = b_hh[2 * H_DIM + j0 + n];
    for (int v = 0; v < 8; ++v) { ahr[v] = br; ahz[v] = bz; ahn[v] = bn; }
  }

  // input gates: gi += x @ Wih^T   (K = 160 for layer0, 1024 for layer1)
  for (int k0 = 0; k0 < Kx; k0 += 32) {
    v16bf a  = load_frag_a(x_bf, Kx, m0, k0, lane);
    v16bf wr = load_frag_b(Wih, Kx, j0, k0, lane);
    v16bf wz = load_frag_b(Wih, Kx, H_DIM + j0, k0, lane);
    v16bf wn = load_frag_b(Wih, Kx, 2 * H_DIM + j0, k0, lane);
    air = wmma_bf16(a, wr, air);
    aiz = wmma_bf16(a, wz, aiz);
    ain = wmma_bf16(a, wn, ain);
  }
  // hidden gates: gh += h @ Whh^T  (K = 1024)
  for (int k0 = 0; k0 < H_DIM; k0 += 32) {
    v16bf a  = load_frag_a(h_bf, H_DIM, m0, k0, lane);
    v16bf wr = load_frag_b(Whh, H_DIM, j0, k0, lane);
    v16bf wz = load_frag_b(Whh, H_DIM, H_DIM + j0, k0, lane);
    v16bf wn = load_frag_b(Whh, H_DIM, 2 * H_DIM + j0, k0, lane);
    ahr = wmma_bf16(a, wr, ahr);
    ahz = wmma_bf16(a, wz, ahz);
    ahn = wmma_bf16(a, wn, ahn);
  }

  // gate combine in registers, write new hidden (f32 + bf16)
  for (int v = 0; v < 8; ++v) {
    size_t idx = (size_t)(mrow + v) * H_DIM + j0 + n;
    float h  = h_f[idx];
    float r  = fsigmoid(air[v] + ahr[v]);
    float z  = fsigmoid(aiz[v] + ahz[v]);
    float nn = ftanh(ain[v] + r * ahn[v]);
    float o  = (1.0f - z) * nn + z * h;
    hnew_f[idx]  = o;
    hnew_bf[idx] = (bf16_t)o;
  }
}

// ---------------- generic GEMM + activation ---------------------------------
// out = act(A @ W^T + bias); optional bf16 mirror. act: 0=relu, 1=sigmoid, 2=none
__global__ __launch_bounds__(256) void gemm_act_kernel(
    const bf16_t* __restrict__ A, int K,
    const bf16_t* __restrict__ W,                   // (>= grid*16 rows) x K
    const float*  __restrict__ bias, int ncols_valid, int act,
    float*  __restrict__ out_f, long long out_ld,
    bf16_t* __restrict__ out_bf, int out_bf_ld)
{
  int lane = threadIdx.x & 31;
  int wave = threadIdx.x >> 5;
  int m0   = wave * 16;
  int j0   = blockIdx.x * 16;
  int n    = lane & 15;
  int col  = j0 + n;

  v8f acc;
  float b = (col < ncols_valid) ? bias[col] : 0.0f;
  for (int v = 0; v < 8; ++v) acc[v] = b;

  for (int k0 = 0; k0 < K; k0 += 32) {
    v16bf a = load_frag_a(A, K, m0, k0, lane);
    v16bf w = load_frag_b(W, K, j0, k0, lane);
    acc = wmma_bf16(a, w, acc);
  }

  int mrow = m0 + ((lane >> 4) << 3);
  for (int v = 0; v < 8; ++v) {
    float x = acc[v];
    if (act == 0)      x = fmaxf(x, 0.0f);
    else if (act == 1) x = fsigmoid(x);
    int r = mrow + v;
    if (col < ncols_valid) out_f[(long long)r * out_ld + col] = x;
    if (out_bf)            out_bf[r * out_bf_ld + col] = (bf16_t)x;
  }
}

// ---------------- fp32 -> bf16 conversion with pad/slice --------------------
__global__ void convert_bf16_kernel(const float* __restrict__ src, int src_rows,
                                    int src_ld, int src_off, int copy_cols,
                                    bf16_t* __restrict__ dst, int dst_rows, int dst_ld)
{
  int idx = blockIdx.x * blockDim.x + threadIdx.x;
  int total = dst_rows * dst_ld;
  if (idx >= total) return;
  int r = idx / dst_ld;
  int c = idx - r * dst_ld;
  float v = 0.0f;
  if (r < src_rows && c < copy_cols) v = src[(size_t)r * src_ld + src_off + c];
  dst[idx] = (bf16_t)v;
}

__global__ void init_decin_kernel(const float* __restrict__ sos,
                                  bf16_t* __restrict__ dec_in)
{
  int idx = blockIdx.x * blockDim.x + threadIdx.x;
  if (idx >= B_DIM * PPAD) return;
  int c = idx % PPAD;
  dec_in[idx] = (bf16_t)((c < P_DIM) ? sos[c] : 0.0f);
}

// ---------------------------------------------------------------------------
extern "C" void kernel_launch(void* const* d_in, const int* in_sizes, int n_in,
                              void* d_out, int out_size, void* d_ws, size_t ws_size,
                              hipStream_t stream)
{
  (void)in_sizes; (void)n_in; (void)out_size; (void)ws_size;
  const float* se    = (const float*)d_in[0];   // (128,768)
  const float* enc_w = (const float*)d_in[1];   // (2048,768)
  const float* enc_b = (const float*)d_in[2];   // (2048)
  const float* w_ih0 = (const float*)d_in[3];   // (3072,918)
  const float* w_hh0 = (const float*)d_in[4];   // (3072,1024)
  const float* b_ih0 = (const float*)d_in[5];
  const float* b_hh0 = (const float*)d_in[6];
  const float* w_ih1 = (const float*)d_in[7];   // (3072,1024)
  const float* w_hh1 = (const float*)d_in[8];
  const float* b_ih1 = (const float*)d_in[9];
  const float* b_hh1 = (const float*)d_in[10];
  const float* out_w = (const float*)d_in[11];  // (150,1024)
  const float* out_b = (const float*)d_in[12];  // (150)
  const float* sos   = (const float*)d_in[13];  // (150)
  float* out = (float*)d_out;                   // (128,256,150)

  // ---- workspace carve-up (~33 MB) ----
  char* ws = (char*)d_ws;
  size_t off = 0;
  auto alloc = [&](size_t bytes) {
    void* p = ws + off;
    off = (off + bytes + 255) & ~(size_t)255;
    return p;
  };
  bf16_t* se_bf   = (bf16_t*)alloc((size_t)B_DIM * E_DIM * 2);
  bf16_t* encw_bf = (bf16_t*)alloc((size_t)2 * H_DIM * E_DIM * 2);
  bf16_t* wih0p   = (bf16_t*)alloc((size_t)G3 * PPAD * 2);   // cols 0..149 (+pad)
  bf16_t* wih0c   = (bf16_t*)alloc((size_t)G3 * E_DIM * 2);  // cols 150..917
  bf16_t* whh0b   = (bf16_t*)alloc((size_t)G3 * H_DIM * 2);
  bf16_t* wih1b   = (bf16_t*)alloc((size_t)G3 * H_DIM * 2);
  bf16_t* whh1b   = (bf16_t*)alloc((size_t)G3 * H_DIM * 2);
  bf16_t* outwb   = (bf16_t*)alloc((size_t)PPAD * H_DIM * 2); // rows 150..159 zero
  float*  gictx   = (float*)alloc((size_t)B_DIM * G3 * 4);
  float*  h0f[2]; bf16_t* h0b[2];
  float*  h1f[2]; bf16_t* h1b[2];
  for (int i = 0; i < 2; ++i) {
    h0f[i] = (float*)alloc((size_t)B_DIM * H_DIM * 4);
    h0b[i] = (bf16_t*)alloc((size_t)B_DIM * H_DIM * 2);
    h1f[i] = (float*)alloc((size_t)B_DIM * H_DIM * 4);
    h1b[i] = (bf16_t*)alloc((size_t)B_DIM * H_DIM * 2);
  }
  bf16_t* decin = (bf16_t*)alloc((size_t)B_DIM * PPAD * 2);

  auto cvt = [&](const float* s, int sr, int sld, int soff, int cc,
                 bf16_t* d, int dr, int dld) {
    int total = dr * dld;
    convert_bf16_kernel<<<dim3((total + 255) / 256), dim3(256), 0, stream>>>(
        s, sr, sld, soff, cc, d, dr, dld);
  };

  // ---- one-time conversions ----
  cvt(se,    B_DIM,    E_DIM, 0,   E_DIM, se_bf,   B_DIM,    E_DIM);
  cvt(enc_w, 2*H_DIM,  E_DIM, 0,   E_DIM, encw_bf, 2*H_DIM,  E_DIM);
  cvt(w_ih0, G3,       918,   0,   P_DIM, wih0p,   G3,       PPAD);   // zero-padded K
  cvt(w_ih0, G3,       918,   P_DIM, E_DIM, wih0c, G3,       E_DIM);
  cvt(w_hh0, G3,       H_DIM, 0,   H_DIM, whh0b,   G3,       H_DIM);
  cvt(w_ih1, G3,       H_DIM, 0,   H_DIM, wih1b,   G3,       H_DIM);
  cvt(w_hh1, G3,       H_DIM, 0,   H_DIM, whh1b,   G3,       H_DIM);
  cvt(out_w, P_DIM,    H_DIM, 0,   H_DIM, outwb,   PPAD,     H_DIM);  // zero-padded rows
  init_decin_kernel<<<dim3((B_DIM*PPAD + 255) / 256), dim3(256), 0, stream>>>(sos, decin);

  // ---- encoder: h_enc = relu(se @ enc_w^T + enc_b), split into h0 | h1 ----
  gemm_act_kernel<<<dim3(H_DIM/16), dim3(256), 0, stream>>>(
      se_bf, E_DIM, encw_bf, enc_b, H_DIM, /*relu*/0,
      h0f[0], H_DIM, h0b[0], H_DIM);
  gemm_act_kernel<<<dim3(H_DIM/16), dim3(256), 0, stream>>>(
      se_bf, E_DIM, encw_bf + (size_t)H_DIM * E_DIM, enc_b + H_DIM, H_DIM, 0,
      h1f[0], H_DIM, h1b[0], H_DIM);

  // ---- gi_ctx = context @ w_ih0[:,150:].T + b_ih0  (loop-invariant) ----
  gemm_act_kernel<<<dim3(G3/16), dim3(256), 0, stream>>>(
      se_bf, E_DIM, wih0c, b_ih0, G3, /*none*/2,
      gictx, G3, (bf16_t*)nullptr, 0);

  // ---- 256-step scan ----
  for (int t = 0; t < LMAX; ++t) {
    int rd = t & 1, wr = rd ^ 1;
    // layer 0: input gates from dec_in (K=160) + gi_ctx; hidden gates from h0
    gru_layer_kernel<<<dim3(H_DIM/16), dim3(256), 0, stream>>>(
        decin, PPAD, wih0p, gictx, (const float*)nullptr,
        h0b[rd], h0f[rd], whh0b, b_hh0,
        h0f[wr], h0b[wr]);
    // layer 1: input = h0_new (K=1024)
    gru_layer_kernel<<<dim3(H_DIM/16), dim3(256), 0, stream>>>(
        h0b[wr], H_DIM, wih1b, (const float*)nullptr, b_ih1,
        h1b[rd], h1f[rd], whh1b, b_hh1,
        h1f[wr], h1b[wr]);
    // pose = sigmoid(h1_new @ out_w^T + out_b) -> d_out[:, t, :] and dec_in
    gemm_act_kernel<<<dim3(PPAD/16), dim3(256), 0, stream>>>(
        h1b[wr], H_DIM, outwb, out_b, P_DIM, /*sigmoid*/1,
        out + (size_t)t * P_DIM, (long long)LMAX * P_DIM,
        decin, PPAD);
  }
}
